// TransformerRadford_24386824307379
// MI455X (gfx1250) — matrix operand order repaired and checked by reading, not compiled, not run on hardware
//
#include <hip/hip_runtime.h>
#include <hip/hip_bf16.h>
#include <math.h>
#include <stdint.h>

typedef __bf16 bf16;
typedef __attribute__((ext_vector_type(16))) __bf16 v16bf;
typedef __attribute__((ext_vector_type(8)))  __bf16 v8bf;
typedef __attribute__((ext_vector_type(8)))  float  v8f;

#define EPI_F32            0
#define EPI_BF16           1
#define EPI_BIAS_RES_F32   2
#define EPI_BIAS_GELU_BF16 3

// ---- CDNA5 data-movement feature detection -------------------------------
#if defined(__has_builtin)
# if __has_builtin(__builtin_amdgcn_tensor_load_to_lds) && \
     __has_builtin(__builtin_amdgcn_s_wait_tensorcnt)
#  define GEMM_TDM 1
# endif
# if __has_builtin(__builtin_amdgcn_global_load_async_to_lds_b128)
#  define GEMM_ASYNC 1
# endif
#endif

#if defined(GEMM_TDM) || defined(GEMM_ASYNC)
# define GEMM_DMA 1
#endif

#define AS1 __attribute__((address_space(1)))
#define AS3 __attribute__((address_space(3)))

__device__ __forceinline__ float gelu_exact(float x) {
  return 0.5f * x * (1.0f + erff(x * 0.70710678118654752f));
}

#ifdef GEMM_TDM
typedef __attribute__((ext_vector_type(4))) unsigned int u32x4;
typedef __attribute__((ext_vector_type(8))) int          i32x8;
typedef __attribute__((ext_vector_type(4))) int          i32x4;

// One TDM descriptor = one 128-row x 32-elem bf16 tile, row stride `ld` elems,
// written row-major contiguous into LDS at byte offset lds_off.
// D# packing per CDNA5 ISA.
// Toolchain here is the 6-arg (clang-23 / therock-10.0) builtin form:
//   (uint32x4 g0, int32x8 g1, int32x4, int32x4, int32x8, i32 cpol)
__device__ __forceinline__ void tdm_load_tile_128x32(const bf16* gsrc,
                                                     unsigned lds_off,
                                                     int ld_elems)
{
  const unsigned long long ga = (unsigned long long)(uintptr_t)gsrc;
  u32x4 g0;
  g0[0] = 1u;                                              // count=1, user mode
  g0[1] = lds_off;                                         // lds_addr (bytes)
  g0[2] = (unsigned)ga;                                    // global_addr[31:0]
  g0[3] = ((unsigned)(ga >> 32) & 0x01FFFFFFu) | (2u << 30); // addr[56:32]|type=2
  i32x8 g1;
  g1[0] = 1 << 16;                 // workgroup_mask=0, data_size=1 (2 bytes)
  g1[1] = (int)0xFFFF0000u;        // atomic=0, tensor_dim0[15:0]=0xFFFF (huge)
  g1[2] = (int)0xFFFF7FFFu;        // tensor_dim0[31:16], tensor_dim1[15:0]
  g1[3] = (32 << 16) | 0x7FFF;     // tensor_dim1[31:16], tile_dim0 = 32
  g1[4] = 128;                     // tile_dim1 = 128, tile_dim2 = 0
  g1[5] = ld_elems;                // tensor_dim0_stride[31:0] (elems)
  g1[6] = 0;                       // stride hi16 | dim1_stride lo16
  g1[7] = 0;
  i32x4 z4 = {0, 0, 0, 0};
  i32x8 z8 = {0, 0, 0, 0, 0, 0, 0, 0};
  __builtin_amdgcn_tensor_load_to_lds(g0, g1, z4, z4, z8, 0);
}
#endif

// ---------------------------------------------------------------------------
// Tiled bf16 WMMA GEMM: C(M x N) = A(M x K, row major) * B(N x K, row major)^T
// 256 threads = 8 wave32, block tile 128x128, wave tile 32x64, K step 32.
// Double-buffered LDS stages filled by TDM (one DMA per tile, wave 0 issues)
// or async global->LDS loads, one barrier per stage.
// ---------------------------------------------------------------------------
template<int EPI>
__global__ __launch_bounds__(256)
void gemm_bf16_wmma(const bf16* __restrict__ Abase, const bf16* __restrict__ Bbase,
                    void* __restrict__ Cbase,
                    const float* __restrict__ bias, const float* __restrict__ resid,
                    int lda, int ldb, int ldc, int K,
                    long aB, long aH, long bB, long bH, long cB, long cH, int Hn)
{
  __shared__ __align__(128) bf16 As[2][128][32];
  __shared__ __align__(128) bf16 Bs[2][128][32];

  const int tid  = threadIdx.x;
  const int wave = tid >> 5;
  const int lane = tid & 31;

  const int  z    = blockIdx.z;
  const int  zb   = z / Hn;
  const int  zh   = z - zb * Hn;
  const long aoff = (long)zb * aB + (long)zh * aH;
  const long boff = (long)zb * bB + (long)zh * bH;
  const long coff = (long)zb * cB + (long)zh * cH;

  const bf16* A = Abase + aoff + (long)blockIdx.y * 128 * lda;
  const bf16* B = Bbase + boff + (long)blockIdx.x * 128 * ldb;

  const int waveM = wave & 3;
  const int waveN = wave >> 2;
  const int mW = waveM * 32;
  const int nW = waveN * 64;

  const int lrow = tid >> 1;          // 0..127
  const int lseg = (tid & 1) * 16;    // 0 or 16

  v8f acc[2][4];
  #pragma unroll
  for (int i = 0; i < 2; ++i)
    #pragma unroll
    for (int j = 0; j < 4; ++j)
      acc[i][j] = v8f{0.f, 0.f, 0.f, 0.f, 0.f, 0.f, 0.f, 0.f};

  const int mL  = lane & 15;
  const int kbA = (lane >> 4) * 8;
  const int kbB = (lane >> 4) * 8;

  const int nStages = K >> 5;

  // ---- stage loaders ------------------------------------------------------
#ifdef GEMM_TDM
  auto dma_issue = [&](int buf, int k0) {
    if (wave == 0) {
      tdm_load_tile_128x32(A + k0, (unsigned)(uintptr_t)&As[buf][0][0], lda);
      tdm_load_tile_128x32(B + k0, (unsigned)(uintptr_t)&Bs[buf][0][0], ldb);
    }
  };
  auto dma_wait = [&]() {
    if (wave == 0) __builtin_amdgcn_s_wait_tensorcnt(0);
  };
#elif defined(GEMM_ASYNC)
  auto dma_issue = [&](int buf, int k0) {
    const bf16* ga0 = &A[(long)lrow * lda + k0 + lseg];
    const bf16* gb0 = &B[(long)lrow * ldb + k0 + lseg];
    __builtin_amdgcn_global_load_async_to_lds_b128(
        (AS1 void*)(void*)const_cast<bf16*>(ga0),
        (AS3 void*)(void*)&As[buf][lrow][lseg], 0, 0);
    __builtin_amdgcn_global_load_async_to_lds_b128(
        (AS1 void*)(void*)const_cast<bf16*>(ga0 + 8),
        (AS3 void*)(void*)&As[buf][lrow][lseg + 8], 0, 0);
    __builtin_amdgcn_global_load_async_to_lds_b128(
        (AS1 void*)(void*)const_cast<bf16*>(gb0),
        (AS3 void*)(void*)&Bs[buf][lrow][lseg], 0, 0);
    __builtin_amdgcn_global_load_async_to_lds_b128(
        (AS1 void*)(void*)const_cast<bf16*>(gb0 + 8),
        (AS3 void*)(void*)&Bs[buf][lrow][lseg + 8], 0, 0);
  };
  auto dma_wait = [&]() {
# if __has_builtin(__builtin_amdgcn_s_wait_asynccnt)
    __builtin_amdgcn_s_wait_asynccnt(0);
# else
    asm volatile("s_wait_asynccnt 0x0" ::: "memory");
# endif
  };
#endif

#ifdef GEMM_DMA
  dma_issue(0, 0);
#endif

  for (int kt = 0; kt < nStages; ++kt) {
    const int buf = kt & 1;
    const int k0  = kt << 5;

#ifdef GEMM_DMA
    dma_wait();            // my stage-kt DMA complete (data in LDS)
    __syncthreads();       // all waves' stage-kt data visible; all done reading buf^1
    if (kt + 1 < nStages) dma_issue(buf ^ 1, k0 + 32);
#else
    __syncthreads();
    *(v8bf*)&As[buf][lrow][lseg]     = *(const v8bf*)&A[(long)lrow * lda + k0 + lseg];
    *(v8bf*)&As[buf][lrow][lseg + 8] = *(const v8bf*)&A[(long)lrow * lda + k0 + lseg + 8];
    *(v8bf*)&Bs[buf][lrow][lseg]     = *(const v8bf*)&B[(long)lrow * ldb + k0 + lseg];
    *(v8bf*)&Bs[buf][lrow][lseg + 8] = *(const v8bf*)&B[(long)lrow * ldb + k0 + lseg + 8];
    __syncthreads();
#endif

    // A fragments (16x32 ISA layout: elems 0..7 -> K=kbA.., 8..15 -> K=16+kbA..)
    v16bf afr[2];
    #pragma unroll
    for (int i = 0; i < 2; ++i) {
      v8bf lo = *(const v8bf*)&As[buf][mW + i * 16 + mL][kbA];
      v8bf hi = *(const v8bf*)&As[buf][mW + i * 16 + mL][16 + kbA];
      afr[i] = __builtin_shufflevector(lo, hi, 0, 1, 2, 3, 4, 5, 6, 7,
                                               8, 9, 10, 11, 12, 13, 14, 15);
    }
    // B fragments (same K order as A: elems 0..7 -> K=kbB.., 8..15 -> K=16+kbB..)
    v16bf bfr[4];
    #pragma unroll
    for (int j = 0; j < 4; ++j) {
      v8bf lo = *(const v8bf*)&Bs[buf][nW + j * 16 + mL][kbB];
      v8bf hi = *(const v8bf*)&Bs[buf][nW + j * 16 + mL][16 + kbB];
      bfr[j] = __builtin_shufflevector(lo, hi, 0, 1, 2, 3, 4, 5, 6, 7,
                                               8, 9, 10, 11, 12, 13, 14, 15);
    }

    #pragma unroll
    for (int i = 0; i < 2; ++i)
      #pragma unroll
      for (int j = 0; j < 4; ++j)
        acc[i][j] = __builtin_amdgcn_wmma_f32_16x16x32_bf16(
            false, afr[i], false, bfr[j], (short)0, acc[i][j], false, false);
  }

  // C/D layout: VGPR r -> M = r + 8*(lane>=16), N = lane%16
  const int crow0 = blockIdx.y * 128 + mW + ((lane >> 4) * 8);
  const int ccol0 = blockIdx.x * 128 + nW + (lane & 15);
  #pragma unroll
  for (int i = 0; i < 2; ++i)
    #pragma unroll
    for (int j = 0; j < 4; ++j)
      #pragma unroll
      for (int r = 0; r < 8; ++r) {
        const long row = crow0 + i * 16 + r;
        const long col = ccol0 + j * 16;
        const long idx = coff + row * (long)ldc + col;
        float vv = acc[i][j][r];
        if (EPI == EPI_F32) {
          ((float*)Cbase)[idx] = vv;
        } else if (EPI == EPI_BF16) {
          ((bf16*)Cbase)[idx] = (bf16)vv;
        } else if (EPI == EPI_BIAS_RES_F32) {
          ((float*)Cbase)[idx] = vv + bias[col] + resid[idx];
        } else { // EPI_BIAS_GELU_BF16
          ((bf16*)Cbase)[idx] = (bf16)gelu_exact(vv + bias[col]);
        }
      }
}

// ---------------------------------------------------------------------------
// LayerNorm over D=4096, one block per token, bf16 output.
// ---------------------------------------------------------------------------
__global__ __launch_bounds__(256)
void ln_to_bf16(const float* __restrict__ x, const float* __restrict__ g,
                const float* __restrict__ bta, bf16* __restrict__ out)
{
  const int  t    = threadIdx.x;
  const long base = (long)blockIdx.x * 4096;
  float s = 0.f, s2 = 0.f;
  for (int i = t; i < 4096; i += 256) {
    float v = x[base + i];
    s += v; s2 += v * v;
  }
  __shared__ float r1[256], r2[256];
  r1[t] = s; r2[t] = s2;
  __syncthreads();
  for (int off = 128; off > 0; off >>= 1) {
    if (t < off) { r1[t] += r1[t + off]; r2[t] += r2[t + off]; }
    __syncthreads();
  }
  const float mean = r1[0] * (1.0f / 4096.0f);
  const float var  = r2[0] * (1.0f / 4096.0f) - mean * mean;
  const float rs   = rsqrtf(var + 1e-5f);
  for (int i = t; i < 4096; i += 256) {
    float v = (x[base + i] - mean) * rs * g[i] + bta[i];
    out[base + i] = (bf16)v;
  }
}

// ---------------------------------------------------------------------------
// mask (tril, recomputed) -> scale (1/sqrt(512)) -> softmax over QUERY axis i.
// s is (B*H, 512, 512); each thread owns a column j; in-place.
// ---------------------------------------------------------------------------
__global__ __launch_bounds__(256)
void softmax_query_axis(float* __restrict__ s)
{
  const int  bh = blockIdx.y;
  const int  j  = blockIdx.x * 256 + threadIdx.x;
  float* sp = s + (long)bh * 512 * 512;
  const float scale = 0.044194173824159216f;   // 1/sqrt(512)
  const float neg   = -1000.0f * scale;

  float m = neg;
  for (int i = 0; i < 512; ++i) {
    float v = (j <= i) ? sp[(long)i * 512 + j] * scale : neg;
    m = fmaxf(m, v);
  }
  float sum = 0.f;
  for (int i = 0; i < 512; ++i) {
    float v = (j <= i) ? sp[(long)i * 512 + j] * scale : neg;
    sum += __expf(v - m);
  }
  const float inv = 1.0f / sum;
  for (int i = 0; i < 512; ++i) {
    float v = (j <= i) ? sp[(long)i * 512 + j] * scale : neg;
    sp[(long)i * 512 + j] = __expf(v - m) * inv;
  }
}

// ---------------------------------------------------------------------------
// att[b, i, h*512 + j] = p[b,h,i,j] * v[b,h,j,i]   (v stored (B,L,H*A))
// ---------------------------------------------------------------------------
__global__ __launch_bounds__(256)
void combine_pv(const float* __restrict__ p, const float* __restrict__ v,
                bf16* __restrict__ att)
{
  const long idx = (long)blockIdx.x * 256 + threadIdx.x;   // over B*L*D
  const int  n  = (int)(idx & 4095);
  const long bi = idx >> 12;
  const int  i  = (int)(bi & 511);
  const int  b  = (int)(bi >> 9);
  const int  h  = n >> 9;
  const int  j  = n & 511;
  const float pv = p[(((long)(b * 8 + h) * 512 + i) * 512) + j];
  const float vv = v[((long)(b * 512 + j) << 12) + h * 512 + i];
  att[idx] = (bf16)(pv * vv);
}

// W (H=8, D=4096, A=512) f32  ->  out (H*A=4096 rows, D=4096 cols) bf16
__global__ __launch_bounds__(256)
void cast_transpose_hda(const float* __restrict__ W, bf16* __restrict__ out)
{
  const long idx = (long)blockIdx.x * 256 + threadIdx.x;   // over 4096*4096
  const int  d   = (int)(idx & 4095);
  const int  row = (int)(idx >> 12);                        // h*512 + a
  const int  h   = row >> 9;
  const int  a   = row & 511;
  out[idx] = (bf16)W[((long)h * 4096 + d) * 512 + a];
}

// W (K x N) f32  ->  out (N x K) bf16
__global__ __launch_bounds__(256)
void cast_transpose_2d(const float* __restrict__ W, bf16* __restrict__ out,
                       int K, int N)
{
  const long idx = (long)blockIdx.x * 256 + threadIdx.x;   // over N*K
  const long n = idx / K;
  const long k = idx - n * K;
  out[idx] = (bf16)W[k * (long)N + n];
}

// ---------------------------------------------------------------------------
extern "C" void kernel_launch(void* const* d_in, const int* in_sizes, int n_in,
                              void* d_out, int out_size, void* d_ws, size_t ws_size,
                              hipStream_t stream)
{
  const float* x    = (const float*)d_in[0];
  /* d_in[1] = mask: deterministic tril, recomputed on device */
  const float* Wq   = (const float*)d_in[2];
  const float* Wk   = (const float*)d_in[3];
  const float* Wv   = (const float*)d_in[4];
  const float* fcw  = (const float*)d_in[5];
  const float* fcb  = (const float*)d_in[6];
  const float* ln1g = (const float*)d_in[7];
  const float* ln1b = (const float*)d_in[8];
  const float* W1   = (const float*)d_in[9];
  const float* b1   = (const float*)d_in[10];
  const float* W2   = (const float*)d_in[11];
  const float* b2   = (const float*)d_in[12];
  const float* ln2g = (const float*)d_in[13];
  const float* ln2b = (const float*)d_in[14];

  const long BL = 4096, Dd = 4096, Ff = 16384, Ll = 512, Hh = 8;

  char*  ws  = (char*)d_ws;
  size_t off = 0;
  auto take = [&](size_t bytes) -> char* {
    char* p = ws + off;
    off = (off + bytes + 255) & ~(size_t)255;
    return p;
  };
  bf16*  xnbf  = (bf16*)take(BL * Dd * 2);
  bf16*  WqT   = (bf16*)take(Dd * Dd * 2);
  bf16*  WkT   = (bf16*)take(Dd * Dd * 2);
  bf16*  WvT   = (bf16*)take(Dd * Dd * 2);
  bf16*  fcwT  = (bf16*)take(Dd * Dd * 2);
  bf16*  W1T   = (bf16*)take(Dd * Ff * 2);
  bf16*  W2T   = (bf16*)take(Ff * Dd * 2);
  bf16*  qbf   = (bf16*)take(BL * Dd * 2);
  bf16*  kbf   = (bf16*)take(BL * Dd * 2);
  float* vf    = (float*)take(BL * Dd * 4);
  float* sbuf  = (float*)take((size_t)64 * Ll * Ll * 4);   // (B*H, L, L)
  bf16*  attbf = (bf16*)take(BL * Dd * 2);
  float* x1    = (float*)take(BL * Dd * 4);
  bf16*  xn2bf = (bf16*)take(BL * Dd * 2);
  bf16*  hbf   = (bf16*)take(BL * Ff * 2);

  const dim3 blk(256);

  // 1) weight casts/transposes to bf16 N x K layout
  cast_transpose_hda<<<65536, blk, 0, stream>>>(Wq, WqT);
  cast_transpose_hda<<<65536, blk, 0, stream>>>(Wk, WkT);
  cast_transpose_hda<<<65536, blk, 0, stream>>>(Wv, WvT);
  cast_transpose_2d <<<65536, blk, 0, stream>>>(fcw, fcwT, 4096, 4096);
  cast_transpose_2d <<<262144, blk, 0, stream>>>(W1, W1T, 4096, 16384);
  cast_transpose_2d <<<262144, blk, 0, stream>>>(W2, W2T, 16384, 4096);

  // 2) LN1
  ln_to_bf16<<<4096, blk, 0, stream>>>(x, ln1g, ln1b, xnbf);

  // 3) q, k (bf16 out), v (f32 out): (4096 x 4096) x (4096 x 4096)
  gemm_bf16_wmma<EPI_BF16><<<dim3(32, 32, 1), blk, 0, stream>>>(
      xnbf, WqT, qbf, nullptr, nullptr, 4096, 4096, 4096, 4096,
      0, 0, 0, 0, 0, 0, 1);
  gemm_bf16_wmma<EPI_BF16><<<dim3(32, 32, 1), blk, 0, stream>>>(
      xnbf, WkT, kbf, nullptr, nullptr, 4096, 4096, 4096, 4096,
      0, 0, 0, 0, 0, 0, 1);
  gemm_bf16_wmma<EPI_F32><<<dim3(32, 32, 1), blk, 0, stream>>>(
      xnbf, WvT, vf, nullptr, nullptr, 4096, 4096, 4096, 4096,
      0, 0, 0, 0, 0, 0, 1);

  // 4) s = q k^T per (b,h): 64 batches of 512 x 512 x 512
  gemm_bf16_wmma<EPI_F32><<<dim3(4, 4, 64), blk, 0, stream>>>(
      qbf, kbf, sbuf, nullptr, nullptr, 4096, 4096, 512, 512,
      (long)Ll * 4096, 512, (long)Ll * 4096, 512,
      (long)Hh * Ll * Ll, (long)Ll * Ll, 8);

  // 5) mask + scale + softmax over query axis
  softmax_query_axis<<<dim3(2, 64), blk, 0, stream>>>(sbuf);

  // 6) att = p * v^T, concat heads -> bf16 (B, L, D)
  combine_pv<<<65536, blk, 0, stream>>>(sbuf, vf, attbf);

  // 7) fc: x1 = att @ fc_w + fc_b + x
  gemm_bf16_wmma<EPI_BIAS_RES_F32><<<dim3(32, 32, 1), blk, 0, stream>>>(
      attbf, fcwT, x1, fcb, x, 4096, 4096, 4096, 4096,
      0, 0, 0, 0, 0, 0, 1);

  // 8) LN2
  ln_to_bf16<<<4096, blk, 0, stream>>>(x1, ln2g, ln2b, xn2bf);

  // 9) MLP up: h = gelu(xn2 @ W1 + b1), bf16 out (4096 x 16384)
  gemm_bf16_wmma<EPI_BIAS_GELU_BF16><<<dim3(128, 32, 1), blk, 0, stream>>>(
      xn2bf, W1T, hbf, b1, nullptr, 4096, 4096, 16384, 4096,
      0, 0, 0, 0, 0, 0, 1);

  // 10) MLP down + residual -> final output (f32)
  gemm_bf16_wmma<EPI_BIAS_RES_F32><<<dim3(32, 32, 1), blk, 0, stream>>>(
      hbf, W2T, d_out, b2, x1, 16384, 16384, 4096, 16384,
      0, 0, 0, 0, 0, 0, 1);
}
